// OptimizedHSI3DTransformer_71210557768285
// MI455X (gfx1250) — matrix operand order repaired and checked by reading, not compiled, not run on hardware
//
#include <hip/hip_runtime.h>
#include <hip/hip_bf16.h>

typedef _Float16 h16;
typedef __attribute__((ext_vector_type(16))) _Float16 v16h;
typedef __attribute__((ext_vector_type(8)))  _Float16 v8h;
typedef __attribute__((ext_vector_type(8)))  float    v8f;

#define DEV __device__ __forceinline__

static constexpr int Bn = 1024, SEQ = 49, ED = 128, HEADS = 4, DH = 32;
static constexpr float EPSV = 1e-5f;

DEV v8f wmma16(v16h a, v16h b, v8f c){
  return __builtin_amdgcn_wmma_f32_16x16x32_f16(false, a, false, b, (short)0, c, false, false);
}

// A fragment, padded LDS: element (m,k) = base[m*stride + k].
// ISA A layout: lane holds M=lane&15; halves 0..7 -> k0+laneHi*8 .. +7,
// halves 8..15 -> k0+16+laneHi*8 .. +7  => two contiguous 16B loads.
DEV v16h ldsAp(const h16* base, int stride, int m0, int k0, int rowcap, int lane){
  int m = m0 + (lane & 15);
  if(m > rowcap) m = rowcap;
  const h16* p = base + m*stride + k0 + ((lane >> 4) << 3);
  v8h lo = *(const v8h*)p;
  v8h hi = *(const v8h*)(p + 16);
  v16h f;
#pragma unroll
  for(int h=0; h<8; ++h){ f[h] = lo[h]; f[h+8] = hi[h]; }
  return f;
}

// A fragment with 7x7 tap remap; OOB taps / pad rows read zero-row 49.
DEV v16h ldsA_convp(const h16* base, int stride, int m0, int k0, int dy, int dx, int lane){
  int m = m0 + (lane & 15);
  int row = 49;
  if(m < SEQ){
    int y = m/7, x = m%7;
    int yy = y+dy, xx = x+dx;
    if((unsigned)yy < 7u && (unsigned)xx < 7u) row = yy*7 + xx;
  }
  const h16* p = base + row*stride + k0 + ((lane >> 4) << 3);
  v8h lo = *(const v8h*)p;
  v8h hi = *(const v8h*)(p + 16);
  v16h f;
#pragma unroll
  for(int h=0; h<8; ++h){ f[h] = lo[h]; f[h+8] = hi[h]; }
  return f;
}

// B fragment, k-contiguous storage [n][kstride]: element (k,n) = base[n*stride + k].
// ISA B layout: lane holds N=lane&15; halves 0..7 -> k0+laneHi*8 .. +7,
// halves 8..15 -> k0+16+laneHi*8 .. +7  => two contiguous 16B loads.
DEV v16h ldsBp(const h16* base, int stride, int k0, int n0, int lane){
  const h16* p = base + (n0 + (lane & 15))*stride + k0 + ((lane >> 4) << 3);
  v8h a = *(const v8h*)p;
  v8h b = *(const v8h*)(p + 16);
  v16h f;
#pragma unroll
  for(int h=0; h<8; ++h){ f[h] = a[h]; f[h+8] = b[h]; }
  return f;
}

// Same pattern for prepped global f16 weights [Npad][Kpad] (zero padded).
DEV v16h gBh(const h16* wt, int Kpad, int n0, int k0, int lane){
  const h16* p = wt + (size_t)(n0 + (lane & 15))*Kpad + k0 + ((lane >> 4) << 3);
  v8h a = *(const v8h*)p;
  v8h b = *(const v8h*)(p + 16);
  v16h f;
#pragma unroll
  for(int h=0; h<8; ++h){ f[h] = a[h]; f[h+8] = b[h]; }
  return f;
}

template<typename F>
DEV void foreach_c(int m0, int n0, int lane, const v8f& c, F&& f){
  int n  = n0 + (lane & 15);
  int mb = m0 + ((lane >> 4) << 3);
#pragma unroll
  for(int r=0; r<8; ++r) f(mb + r, n, c[r]);
}

DEV float geluf(float x){ return 0.5f*x*(1.f + erff(x*0.70710678118654752f)); }
DEV float sigm(float x){ return 1.f/(1.f + expf(-x)); }
DEV float clip10(float x){ return fminf(10.f, fmaxf(-10.f, x)); }

// ------------------------------------------------------- weight prep kernel
// mode 0: linear w[K][N]  src[k*N+n] ; mode 1: conv OIHW src[(n*K+k)*KK+tap]
// dst[(tap*Npad + n)*Kpad + k], zero padded.
__global__ void k_wprep(const float* __restrict__ src, h16* __restrict__ dst,
                        int K, int N, int KK, int Kpad, int Npad, int mode){
  int total = KK*Npad*Kpad;
  for(int i = blockIdx.x*blockDim.x + threadIdx.x; i < total; i += gridDim.x*blockDim.x){
    int tap = i/(Npad*Kpad);
    int r = i%(Npad*Kpad);
    int n = r/Kpad, k = r%Kpad;
    float v = 0.f;
    if(n < N && k < K) v = (mode == 0) ? src[k*N + n] : src[(n*K + k)*KK + tap];
    dst[i] = (h16)v;
  }
}

// ---------------------------------------------------------------- PCA kernel
__global__ void k_pca(const float* __restrict__ x, const float* __restrict__ bnb,
                      const float* __restrict__ bng, const float* __restrict__ mean,
                      const float* __restrict__ comp, float* __restrict__ xpca){
  __shared__ float sx[SEQ*200];
  int b = blockIdx.x, tid = threadIdx.x;
  const float s0 = rsqrtf(1.f + EPSV);
  const float* xb = x + (size_t)b*200*SEQ;
  for(int i=tid; i<200*SEQ; i+=256){
    int c = i/SEQ, p = i%SEQ;
    float v = xb[i]*bng[c]*s0 + bnb[c];
    sx[p*200 + c] = clip10(v) - mean[c];
  }
  __syncthreads();
  for(int i=tid; i<SEQ*30; i+=256){
    int p = i/30, j = i%30;
    const float* cr = comp + j*200;
    const float* xr = sx + p*200;
    float acc = 0.f;
    for(int c=0; c<200; ++c) acc += xr[c]*cr[c];
    xpca[(size_t)b*SEQ*30 + i] = clip10(acc);
  }
}

// ------------------------------------------------- conv0 (3x3, implicit GEMM)
__global__ void k_conv0(const float* __restrict__ in, int cin, int chw_bn, int Kpad,
                        const float* __restrict__ ing, const float* __restrict__ inb,
                        const h16* __restrict__ wt, const float* __restrict__ cb,
                        const float* __restrict__ g, const float* __restrict__ bb,
                        float* __restrict__ out){
  __shared__ __align__(16) h16 sin[50*224];
  int b = blockIdx.x, tid = threadIdx.x;
  int lane = tid & 31, wid = tid >> 5;
  const float s0 = rsqrtf(1.f + EPSV);
  for(int i=tid; i<50*224; i+=256) sin[i] = (h16)0.f;
  __syncthreads();
  if(chw_bn){
    const float* xb = in + (size_t)b*cin*SEQ;
    for(int i=tid; i<cin*SEQ; i+=256){
      int c = i/SEQ, p = i%SEQ;
      float v = clip10(xb[i]*ing[c]*s0 + inb[c]);
      sin[p*224 + c] = (h16)v;
    }
  } else {
    const float* xb = in + (size_t)b*SEQ*cin;
    for(int i=tid; i<SEQ*cin; i+=256){
      int p = i/cin, c = i%cin;
      sin[p*224 + c] = (h16)xb[i];
    }
  }
  __syncthreads();
  for(int t=wid; t<32; t+=8){
    int mt = t & 3, nt = t >> 2;
    v8f acc = {};
    for(int tap=0; tap<9; ++tap){
      int dy = tap/3 - 1, dx = tap%3 - 1;
      const h16* wtt = wt + (size_t)tap*128*Kpad;
      for(int k0=0; k0<Kpad; k0+=32){
        v16h A  = ldsA_convp(sin, 224, mt*16, k0, dy, dx, lane);
        v16h Bf = gBh(wtt, Kpad, nt*16, k0, lane);
        acc = wmma16(A, Bf, acc);
      }
    }
    float* ob = out + (size_t)b*SEQ*ED;
    foreach_c(mt*16, nt*16, lane, acc, [&](int m, int n, float v){
      if(m < SEQ){
        float y = (v + cb[n])*g[n]*s0 + bb[n];
        ob[m*ED + n] = fmaxf(y, 0.f);
      }
    });
  }
}

// ------------------------------------------------------------ msconv kernel
__global__ void k_ms(const float* __restrict__ in,
                     const h16* __restrict__ c1t, const float* __restrict__ c1b,
                     const float* __restrict__ g1, const float* __restrict__ b1,
                     const h16* __restrict__ c3t, const float* __restrict__ c3b,
                     const float* __restrict__ g3, const float* __restrict__ b3,
                     const h16* __restrict__ c5t, const float* __restrict__ c5b,
                     const float* __restrict__ g5, const float* __restrict__ b5,
                     const h16* __restrict__ pjt, const float* __restrict__ pb,
                     const float* __restrict__ gp, const float* __restrict__ bp,
                     float* __restrict__ out){
  __shared__ __align__(16) h16 buf[50*256]; // cols: x[0,128) o1[128,170) o2[170,212) o3[212,254)
  int b = blockIdx.x, tid = threadIdx.x;
  int lane = tid & 31, wid = tid >> 5;
  const float s0 = rsqrtf(1.f + EPSV);
  for(int i=tid; i<50*256; i+=256) buf[i] = (h16)0.f;
  __syncthreads();
  const float* ib = in + (size_t)b*SEQ*ED;
  for(int i=tid; i<SEQ*ED; i+=256){
    int p = i/ED, c = i%ED;
    buf[p*256 + c] = (h16)ib[i];
  }
  __syncthreads();
  // c1: 1x1, K=128 -> 42
  for(int t=wid; t<12; t+=8){
    int mt = t & 3, nt = t >> 2;
    v8f acc = {};
    for(int k0=0; k0<128; k0+=32){
      v16h A  = ldsAp(buf, 256, mt*16, k0, 49, lane);
      v16h Bf = gBh(c1t, 128, nt*16, k0, lane);
      acc = wmma16(A, Bf, acc);
    }
    foreach_c(mt*16, nt*16, lane, acc, [&](int m, int n, float v){
      if(m < SEQ && n < 42){
        float y = (v + c1b[n])*g1[n]*s0 + b1[n];
        buf[m*256 + 128 + n] = (h16)fmaxf(y, 0.f);
      }
    });
  }
  __syncthreads();
  // c3: 3x3, K=170 (Kpad 192, weight zero-padded) -> 42
  for(int t=wid; t<12; t+=8){
    int mt = t & 3, nt = t >> 2;
    v8f acc = {};
    for(int tap=0; tap<9; ++tap){
      int dy = tap/3 - 1, dx = tap%3 - 1;
      const h16* wtt = c3t + (size_t)tap*48*192;
      for(int k0=0; k0<192; k0+=32){
        v16h A  = ldsA_convp(buf, 256, mt*16, k0, dy, dx, lane);
        v16h Bf = gBh(wtt, 192, nt*16, k0, lane);
        acc = wmma16(A, Bf, acc);
      }
    }
    foreach_c(mt*16, nt*16, lane, acc, [&](int m, int n, float v){
      if(m < SEQ && n < 42){
        float y = (v + c3b[n])*g3[n]*s0 + b3[n];
        buf[m*256 + 170 + n] = (h16)fmaxf(y, 0.f);
      }
    });
  }
  __syncthreads();
  // c5: 5x5, K=212 (Kpad 224) -> 42
  for(int t=wid; t<12; t+=8){
    int mt = t & 3, nt = t >> 2;
    v8f acc = {};
    for(int tap=0; tap<25; ++tap){
      int dy = tap/5 - 2, dx = tap%5 - 2;
      const h16* wtt = c5t + (size_t)tap*48*224;
      for(int k0=0; k0<224; k0+=32){
        v16h A  = ldsA_convp(buf, 256, mt*16, k0, dy, dx, lane);
        v16h Bf = gBh(wtt, 224, nt*16, k0, lane);
        acc = wmma16(A, Bf, acc);
      }
    }
    foreach_c(mt*16, nt*16, lane, acc, [&](int m, int n, float v){
      if(m < SEQ && n < 42){
        float y = (v + c5b[n])*g5[n]*s0 + b5[n];
        buf[m*256 + 212 + n] = (h16)fmaxf(y, 0.f);
      }
    });
  }
  __syncthreads();
  // proj: 1x1 on concat(o1,o2,o3), K=126 (Kpad 128) -> 128
  for(int t=wid; t<32; t+=8){
    int mt = t & 3, nt = t >> 2;
    v8f acc = {};
    for(int k0=0; k0<128; k0+=32){
      v16h A  = ldsAp(buf + 128, 256, mt*16, k0, 49, lane);
      v16h Bf = gBh(pjt, 128, nt*16, k0, lane);
      acc = wmma16(A, Bf, acc);
    }
    float* ob = out + (size_t)b*SEQ*ED;
    foreach_c(mt*16, nt*16, lane, acc, [&](int m, int n, float v){
      if(m < SEQ){
        float y = (v + pb[n])*gp[n]*s0 + bp[n];
        ob[m*ED + n] = fmaxf(y, 0.f);
      }
    });
  }
}

// -------------------------------------------------------- transformer kernel
struct LayerP {
  const h16 *qkvt, *outwt, *f1t, *f2t;
  const float *outb, *f1b, *f2b;
  const float *sc1w, *sc1b, *sbn1g, *sbn1b;
  const float *sc2w, *sc2b, *sbn2g, *sbn2b;
  const float *ca1w, *ca1b, *ca2w, *ca2b;
  const float *ln1g, *ln1b, *ln2g, *ln2b;
};
struct TFP { const float *lnb, *lng, *pos; LayerP L[2]; };

// scr layout (halfs)
static constexpr int OQ  = 0;      // qbuf  50*32
static constexpr int OKB = 1600;   // kbuf  64*32 (B-side, n<=63 reads)
static constexpr int OT3 = 3648;   // t3    50*64 (softmax out, zero-padded cols)
static constexpr int OVT = 6848;   // vT    32*64 (v transposed, zero-padded)
static constexpr int OT4 = 8896;   // t4    50*32 (per-head attn out)
static constexpr int OT1 = 10496;  // t1    51*52 halo map
static constexpr int OT2 = 13152;  // t2    51*52 halo map
static constexpr int SCRH = 15808;

DEV void ln_rows(const h16* src, h16* dst, const float* g, const float* b, int tid){
  if(tid < SEQ){
    const h16* sr = src + tid*ED;
    float s = 0.f, s2 = 0.f;
    for(int c8=0; c8<16; ++c8){
      v8h v = *(const v8h*)(sr + c8*8);
#pragma unroll
      for(int j=0; j<8; ++j){ float x = (float)v[j]; s += x; s2 += x*x; }
    }
    float m = s/128.f, inv = rsqrtf(s2/128.f - m*m + EPSV);
    h16* dr = dst + tid*ED;
    for(int c8=0; c8<16; ++c8){
      v8h v = *(const v8h*)(sr + c8*8);
      v8h o;
#pragma unroll
      for(int j=0; j<8; ++j){
        int c = c8*8 + j;
        o[j] = (h16)(((float)v[j] - m)*inv*g[c] + b[c]);
      }
      *(v8h*)(dr + c8*8) = o;
    }
  }
}

DEV void dwconv49h(const h16* src, h16* dst, const float* w9, const float* cb,
                   const float* g, const float* bb, int hidx, int tid){
  float w[9];
#pragma unroll
  for(int i=0; i<9; ++i) w[i] = w9[hidx*9 + i];
  float bias = cb[hidx];
  float sc = g[hidx]*rsqrtf(1.f + EPSV);
  float sb = bb[hidx];
  for(int i=tid; i<SEQ*SEQ; i+=256){
    int y = i/SEQ + 1, x = i%SEQ + 1;
    const h16* s = src + y*52 + x;
    float a = bias;
    a += (float)s[-53]*w[0] + (float)s[-52]*w[1] + (float)s[-51]*w[2];
    a += (float)s[-1] *w[3] + (float)s[0]  *w[4] + (float)s[1]  *w[5];
    a += (float)s[51] *w[6] + (float)s[52] *w[7] + (float)s[53] *w[8];
    dst[y*52 + x] = (h16)fmaxf(a*sc + sb, 0.f);
  }
}

// Builds post-conv attention map for head hidx into t1 (halo layout)
__device__ void head_attnmap(const h16* hbuf, h16* scr, const LayerP& Lp, int hidx, int tid){
  int lane = tid & 31, wid = tid >> 5;
  h16* qbuf = scr + OQ;
  h16* kbuf = scr + OKB;
  h16* t1   = scr + OT1;
  h16* t2   = scr + OT2;
  // q,k GEMM: M=49 K=128 N=64 (q cols | k cols)
  for(int t=wid; t<16; t+=8){
    int mt = t & 3, nt = t >> 2;
    int n0 = (nt < 2) ? (hidx*DH + nt*16) : (ED + hidx*DH + (nt-2)*16);
    v8f acc = {};
    for(int k0=0; k0<ED; k0+=32){
      v16h A  = ldsAp(hbuf, ED, mt*16, k0, 49, lane);
      v16h Bf = gBh(Lp.qkvt, 128, n0, k0, lane);
      acc = wmma16(A, Bf, acc);
    }
    h16* dst = (nt < 2) ? qbuf : kbuf;
    int nb = (nt & 1)*16;
    foreach_c(mt*16, nb, lane, acc, [&](int m, int n, float v){
      if(m < SEQ) dst[m*DH + n] = (h16)v;
    });
  }
  __syncthreads();
  // dots = q @ k^T * scale : M=49 N=49 K=32  -> t1 halo interior
  for(int t=wid; t<16; t+=8){
    int mt = t & 3, nt = t >> 2;
    v8f acc = {};
    v16h A  = ldsAp(qbuf, DH, mt*16, 0, 49, lane);
    v16h Bf = ldsBp(kbuf, DH, 0, nt*16, lane);
    acc = wmma16(A, Bf, acc);
    foreach_c(mt*16, nt*16, lane, acc, [&](int m, int n, float v){
      if(m < SEQ && n < SEQ) t1[(m+1)*52 + n + 1] = (h16)(v*0.17677669529663687f);
    });
  }
  __syncthreads();
  dwconv49h(t1, t2, Lp.sc1w, Lp.sc1b, Lp.sbn1g, Lp.sbn1b, hidx, tid);
  __syncthreads();
  dwconv49h(t2, t1, Lp.sc2w, Lp.sc2b, Lp.sbn2g, Lp.sbn2b, hidx, tid);
  __syncthreads();
}

__global__ void k_tf(const float* __restrict__ feat, TFP P, int spatial,
                     float* __restrict__ pooled){
  __shared__ __align__(16) h16 xres[SEQ*ED];
  __shared__ __align__(16) h16 hbuf[50*ED];
  __shared__ __align__(16) h16 scr[SCRH];
  __shared__ float sps[8];   // pooled[0..3], cw[4..7]
  __shared__ float red8[8];

  int b = blockIdx.x, tid = threadIdx.x;
  int lane = tid & 31, wid = tid >> 5;

  for(int i=tid; i<50*ED; i+=256) hbuf[i] = (h16)0.f;
  for(int i=tid; i<SCRH; i+=256)  scr[i]  = (h16)0.f;

  // entry: LN over channels + positional embedding
  if(tid < SEQ){
    const float* fr = feat + (size_t)b*SEQ*ED + tid*ED;
    float s = 0.f, s2 = 0.f;
    for(int c=0; c<ED; ++c){ float v = fr[c]; s += v; s2 += v*v; }
    float m = s/(float)ED;
    float inv = rsqrtf(s2/(float)ED - m*m + EPSV);
    for(int c=0; c<ED; ++c){
      float v = (fr[c]-m)*inv*P.lng[c] + P.lnb[c] + P.pos[tid*ED + c];
      xres[tid*ED + c] = (h16)v;
    }
  }
  __syncthreads();

  h16* t1 = scr + OT1;
  h16* t3 = scr + OT3;
  h16* vT = scr + OVT;
  h16* t4 = scr + OT4;

  for(int l=0; l<2; ++l){
    const LayerP& Lp = P.L[l];
    ln_rows(xres, hbuf, Lp.ln1g, Lp.ln1b, tid);
    for(int i=tid; i<2*2656; i+=256) scr[OT1 + i] = (h16)0.f;  // reset halo maps
    __syncthreads();

    if(spatial){
      for(int h=0; h<HEADS; ++h){
        head_attnmap(hbuf, scr, Lp, h, tid);
        float part = 0.f;
        for(int i=tid; i<SEQ*SEQ; i+=256)
          part += (float)t1[(i/SEQ + 1)*52 + i%SEQ + 1];
        part += __shfl_xor(part, 16);
        part += __shfl_xor(part, 8);
        part += __shfl_xor(part, 4);
        part += __shfl_xor(part, 2);
        part += __shfl_xor(part, 1);
        if(lane == 0) red8[wid] = part;
        __syncthreads();
        if(tid == 0){
          float s = 0.f;
          for(int wv=0; wv<8; ++wv) s += red8[wv];
          sps[h] = s/2401.f;
        }
        __syncthreads();
      }
      if(tid == 0){
        float hid[2];
        for(int c2=0; c2<2; ++c2){
          float a = Lp.ca1b[c2];
          for(int h=0; h<4; ++h) a += sps[h]*Lp.ca1w[c2*4 + h];
          hid[c2] = fmaxf(a, 0.f);
        }
        for(int h=0; h<4; ++h){
          float a = Lp.ca2b[h];
          for(int c2=0; c2<2; ++c2) a += hid[c2]*Lp.ca2w[h*2 + c2];
          sps[4+h] = sigm(a);
        }
      }
    } else {
      if(tid < 4) sps[4+tid] = 1.f;
    }
    __syncthreads();

    // output projection accumulated per head in registers (K=32 slices)
    v8f pacc[4];
#pragma unroll
    for(int i=0; i<4; ++i) pacc[i] = (v8f){};

    for(int h=0; h<HEADS; ++h){
      head_attnmap(hbuf, scr, Lp, h, tid);
      // softmax over j with channel weight -> t3 [49][64] zero padded
      if(tid < SEQ){
        float cwh = sps[4+h];
        const h16* row = t1 + (tid+1)*52 + 1;
        float mx = -1e30f;
        for(int j=0; j<SEQ; ++j) mx = fmaxf(mx, (float)row[j]*cwh);
        float sum = 0.f;
        for(int j=0; j<SEQ; ++j) sum += expf((float)row[j]*cwh - mx);
        float inv = 1.f/sum;
        h16* orow = t3 + tid*64;
        for(int j=0; j<SEQ; ++j) orow[j] = (h16)(expf((float)row[j]*cwh - mx)*inv);
        for(int j=SEQ; j<64; ++j) orow[j] = (h16)0.f;
      }
      __syncthreads();
      // v GEMM -> vT[ch][token] (transposed, zero padded tokens)
      for(int t=wid; t<8; t+=8){
        int mt = t & 3, nt = t >> 2;
        v8f acc = {};
        for(int k0=0; k0<ED; k0+=32){
          v16h A  = ldsAp(hbuf, ED, mt*16, k0, 49, lane);
          v16h Bf = gBh(Lp.qkvt, 128, 2*ED + h*DH + nt*16, k0, lane);
          acc = wmma16(A, Bf, acc);
        }
        foreach_c(mt*16, nt*16, lane, acc, [&](int m, int n, float v){
          vT[n*64 + m] = (h16)((m < SEQ) ? v : 0.f);
        });
      }
      __syncthreads();
      // out_h = attn @ v : M=49 K=64(zero pad) N=32 -> t4
      for(int t=wid; t<8; t+=8){
        int mt = t & 3, nt = t >> 2;
        v8f acc = {};
        for(int k0=0; k0<64; k0+=32){
          v16h A  = ldsAp(t3, 64, mt*16, k0, 49, lane);
          v16h Bf = ldsBp(vT, 64, k0, nt*16, lane);
          acc = wmma16(A, Bf, acc);
        }
        foreach_c(mt*16, nt*16, lane, acc, [&](int m, int n, float v){
          if(m < SEQ) t4[m*DH + n] = (h16)v;
        });
      }
      __syncthreads();
      // partial output projection: += t4 @ Wout[h*32:(h+1)*32, :]
#pragma unroll
      for(int i=0; i<4; ++i){
        int tile = wid + 8*i;
        int mt = tile & 3, nt = tile >> 2;
        v16h A  = ldsAp(t4, DH, mt*16, 0, 49, lane);
        v16h Bf = gBh(Lp.outwt, 128, nt*16, h*DH, lane);
        pacc[i] = wmma16(A, Bf, pacc[i]);
      }
      __syncthreads();
    }

    // residual add of projected attention output
#pragma unroll
    for(int i=0; i<4; ++i){
      int tile = wid + 8*i;
      int mt = tile & 3, nt = tile >> 2;
      foreach_c(mt*16, nt*16, lane, pacc[i], [&](int m, int n, float v){
        if(m < SEQ) xres[m*ED + n] = (h16)((float)xres[m*ED + n] + v + Lp.outb[n]);
      });
    }
    __syncthreads();

    ln_rows(xres, hbuf, Lp.ln2g, Lp.ln2b, tid);
    __syncthreads();

    // FFN1: 128 -> 256 (exact gelu), ffnh aliases scr[0..]
    h16* ffnh = scr;
    for(int t=wid; t<64; t+=8){
      int mt = t & 3, nt = t >> 2;
      v8f acc = {};
      for(int k0=0; k0<ED; k0+=32){
        v16h A  = ldsAp(hbuf, ED, mt*16, k0, 49, lane);
        v16h Bf = gBh(Lp.f1t, 128, nt*16, k0, lane);
        acc = wmma16(A, Bf, acc);
      }
      foreach_c(mt*16, nt*16, lane, acc, [&](int m, int n, float v){
        if(m < SEQ) ffnh[m*256 + n] = (h16)geluf(v + Lp.f1b[n]);
      });
    }
    __syncthreads();
    // FFN2: 256 -> 128 + residual
    for(int t=wid; t<32; t+=8){
      int mt = t & 3, nt = t >> 2;
      v8f acc = {};
      for(int k0=0; k0<256; k0+=32){
        v16h A  = ldsAp(ffnh, 256, mt*16, k0, 49, lane);
        v16h Bf = gBh(Lp.f2t, 256, nt*16, k0, lane);
        acc = wmma16(A, Bf, acc);
      }
      foreach_c(mt*16, nt*16, lane, acc, [&](int m, int n, float v){
        if(m < SEQ) xres[m*ED + n] = (h16)((float)xres[m*ED + n] + v + Lp.f2b[n]);
      });
    }
    __syncthreads();
  }

  // mean / max pooling over tokens
  if(tid < ED){
    float mn = 0.f, mx = -1e30f;
    for(int m=0; m<SEQ; ++m){
      float v = (float)xres[m*ED + tid];
      mn += v; mx = fmaxf(mx, v);
    }
    pooled[(size_t)b*256 + tid]      = mn/(float)SEQ;
    pooled[(size_t)b*256 + ED + tid] = mx;
  }
}

// --------------------------------------------------------------- head kernel
__global__ void k_head(const float* __restrict__ pspa, const float* __restrict__ pspe,
                       const h16* __restrict__ f1t, const float* __restrict__ f1b,
                       const float* __restrict__ l1g, const float* __restrict__ l1b,
                       const h16* __restrict__ f2t, const float* __restrict__ f2b,
                       const float* __restrict__ l2g, const float* __restrict__ l2b,
                       const h16* __restrict__ c1t, const float* __restrict__ c1b,
                       const h16* __restrict__ c2t, const float* __restrict__ c2b,
                       float* __restrict__ outp){
  __shared__ __align__(16) h16 fused[32*512];
  __shared__ __align__(16) h16 h1[32*256];
  __shared__ __align__(16) h16 h2[32*128];
  __shared__ __align__(16) h16 h3[32*64];
  int tid = threadIdx.x, lane = tid & 31, wid = tid >> 5;
  int img0 = blockIdx.x*32;

  if(tid < 32){
    const float* ps = pspa + (size_t)(img0 + tid)*256;
    const float* pe = pspe + (size_t)(img0 + tid)*256;
    float na = 0.f, ne = 0.f;
    for(int c=0; c<256; ++c){ na += ps[c]*ps[c]; ne += pe[c]*pe[c]; }
    float ia = 1.f/fmaxf(sqrtf(na), 1e-12f);
    float ie = 1.f/fmaxf(sqrtf(ne), 1e-12f);
    for(int c=0; c<256; ++c){
      fused[tid*512 + c]       = (h16)(ps[c]*ia);
      fused[tid*512 + 256 + c] = (h16)(pe[c]*ie);
    }
  }
  __syncthreads();
  // fus1: 512 -> 256
  for(int t=wid; t<32; t+=8){
    int mt = t & 1, nt = t >> 1;
    v8f acc = {};
    for(int k0=0; k0<512; k0+=32){
      v16h A  = ldsAp(fused, 512, mt*16, k0, 31, lane);
      v16h Bf = gBh(f1t, 512, nt*16, k0, lane);
      acc = wmma16(A, Bf, acc);
    }
    foreach_c(mt*16, nt*16, lane, acc, [&](int m, int n, float v){
      h1[m*256 + n] = (h16)(v + f1b[n]);
    });
  }
  __syncthreads();
  if(tid < 32){
    h16* r = h1 + tid*256;
    float s = 0.f, s2 = 0.f;
    for(int c=0; c<256; ++c){ float v = (float)r[c]; s += v; s2 += v*v; }
    float m = s/256.f, inv = rsqrtf(s2/256.f - m*m + EPSV);
    for(int c=0; c<256; ++c){
      float v = ((float)r[c]-m)*inv*l1g[c] + l1b[c];
      r[c] = (h16)fmaxf(v, 0.f);
    }
  }
  __syncthreads();
  // fus2: 256 -> 128
  for(int t=wid; t<16; t+=8){
    int mt = t & 1, nt = t >> 1;
    v8f acc = {};
    for(int k0=0; k0<256; k0+=32){
      v16h A  = ldsAp(h1, 256, mt*16, k0, 31, lane);
      v16h Bf = gBh(f2t, 256, nt*16, k0, lane);
      acc = wmma16(A, Bf, acc);
    }
    foreach_c(mt*16, nt*16, lane, acc, [&](int m, int n, float v){
      h2[m*128 + n] = (h16)(v + f2b[n]);
    });
  }
  __syncthreads();
  if(tid < 32){
    h16* r = h2 + tid*128;
    float s = 0.f, s2 = 0.f;
    for(int c=0; c<128; ++c){ float v = (float)r[c]; s += v; s2 += v*v; }
    float m = s/128.f, inv = rsqrtf(s2/128.f - m*m + EPSV);
    for(int c=0; c<128; ++c){
      float v = ((float)r[c]-m)*inv*l2g[c] + l2b[c];
      r[c] = (h16)fmaxf(v, 0.f);
    }
  }
  __syncthreads();
  // cls1: 128 -> 64 (relu)
  for(int t=wid; t<8; t+=8){
    int mt = t & 1, nt = t >> 1;
    v8f acc = {};
    for(int k0=0; k0<128; k0+=32){
      v16h A  = ldsAp(h2, 128, mt*16, k0, 31, lane);
      v16h Bf = gBh(c1t, 128, nt*16, k0, lane);
      acc = wmma16(A, Bf, acc);
    }
    foreach_c(mt*16, nt*16, lane, acc, [&](int m, int n, float v){
      h3[m*64 + n] = (h16)fmaxf(v + c1b[n], 0.f);
    });
  }
  __syncthreads();
  // cls2: 64 -> 16
  for(int t=wid; t<2; t+=8){
    int mt = t;
    v8f acc = {};
    for(int k0=0; k0<64; k0+=32){
      v16h A  = ldsAp(h3, 64, mt*16, k0, 31, lane);
      v16h Bf = gBh(c2t, 64, 0, k0, lane);
      acc = wmma16(A, Bf, acc);
    }
    foreach_c(mt*16, 0, lane, acc, [&](int m, int n, float v){
      outp[(size_t)(img0 + m)*16 + n] = v + c2b[n];
    });
  }
}

// ----------------------------------------------------------------- launcher
extern "C" void kernel_launch(void* const* d_in, const int* in_sizes, int n_in,
                              void* d_out, int out_size, void* d_ws, size_t ws_size,
                              hipStream_t stream){
  auto F = [&](int i){ return (const float*)d_in[i]; };
  const float* X = F(0);

  float* ws = (float*)d_ws;
  size_t o = 0;
  float* xpca  = ws + o; o += (size_t)Bn*SEQ*30;
  float* featA = ws + o; o += (size_t)Bn*SEQ*ED;
  float* featB = ws + o; o += (size_t)Bn*SEQ*ED;
  float* pspa  = ws + o; o += (size_t)Bn*256;
  float* pspe  = ws + o; o += (size_t)Bn*256;

  h16* hp = (h16*)(ws + o);   // f16 weight pool (offset is 16B aligned)
  auto prep = [&](int srcIdx, int K, int N, int KK, int Kpad, int Npad, int mode)->const h16*{
    h16* dst = hp;
    size_t sz = (size_t)KK*Npad*Kpad;
    hp += sz;
    int total = (int)sz;
    int gridn = (total + 255)/256; if(gridn > 2048) gridn = 2048;
    k_wprep<<<gridn, 256, 0, stream>>>(F(srcIdx), dst, K, N, KK, Kpad, Npad, mode);
    return dst;
  };

  // conv weights (mode 1, OIHW)
  const h16* spa0t = prep(20, 30, 128, 9, 32, 128, 1);
  const h16* spe0t = prep(89, 200, 128, 9, 224, 128, 1);
  const h16* sa_c1 = prep(32, 128, 42, 1, 128, 48, 1);
  const h16* sa_c3 = prep(34, 170, 42, 9, 192, 48, 1);
  const h16* sa_c5 = prep(36, 212, 42, 25, 224, 48, 1);
  const h16* sa_pj = prep(38, 126, 128, 1, 128, 128, 1);
  const h16* se_c1 = prep(101, 128, 42, 1, 128, 48, 1);
  const h16* se_c3 = prep(103, 170, 42, 9, 192, 48, 1);
  const h16* se_c5 = prep(105, 212, 42, 25, 224, 48, 1);
  const h16* se_pj = prep(107, 126, 128, 1, 128, 128, 1);
  // linear weights (mode 0)
  const h16* qk_t[4]; const h16* ow_t[4]; const h16* w1_t[4]; const h16* w2_t[4];
  int lbase[4] = {40, 63, 109, 132};
  for(int i=0; i<4; ++i){
    qk_t[i] = prep(lbase[i]+6,  128, 384, 1, 128, 384, 0);
    ow_t[i] = prep(lbase[i]+5,  128, 128, 1, 128, 128, 0);
    w1_t[i] = prep(lbase[i]+16, 128, 256, 1, 128, 256, 0);
    w2_t[i] = prep(lbase[i]+18, 256, 128, 1, 256, 128, 0);
  }
  const h16* fu1t = prep(10, 512, 256, 1, 512, 256, 0);
  const h16* fu2t = prep(12, 256, 128, 1, 256, 128, 0);
  const h16* cl1t = prep(2,  128, 64,  1, 128, 64,  0);
  const h16* cl2t = prep(4,  64,  16,  1, 64,  16,  0);

  auto mkLayer = [&](int bidx, int li)->LayerP{
    LayerP L;
    L.qkvt = qk_t[li]; L.outwt = ow_t[li]; L.f1t = w1_t[li]; L.f2t = w2_t[li];
    L.ca1b = F(bidx+0);  L.ca1w = F(bidx+1);  L.ca2b = F(bidx+2);  L.ca2w = F(bidx+3);
    L.outb = F(bidx+4);
    L.sbn1b= F(bidx+7);  L.sbn1g= F(bidx+8);  L.sbn2b= F(bidx+9);  L.sbn2g= F(bidx+10);
    L.sc1b = F(bidx+11); L.sc1w = F(bidx+12); L.sc2b = F(bidx+13); L.sc2w = F(bidx+14);
    L.f1b  = F(bidx+15); L.f2b  = F(bidx+17);
    L.ln1b = F(bidx+19); L.ln1g = F(bidx+20); L.ln2b = F(bidx+21); L.ln2g = F(bidx+22);
    return L;
  };

  // PCA (in_bn: b=13 g=14; pca_comp=15 pca_mean=16)
  k_pca<<<Bn, 256, 0, stream>>>(X, F(13), F(14), F(16), F(15), xpca);

  // ---- spatial branch ----
  k_conv0<<<Bn, 256, 0, stream>>>(xpca, 30, 0, 32, nullptr, nullptr,
                                  spa0t, F(19), F(18), F(17), featA);
  k_ms<<<Bn, 256, 0, stream>>>(featA,
      sa_c1, F(31), F(24), F(23),
      sa_c3, F(33), F(26), F(25),
      sa_c5, F(35), F(28), F(27),
      sa_pj, F(37), F(30), F(29),
      featB);
  TFP spa; spa.lnb = F(21); spa.lng = F(22); spa.pos = F(39);
  spa.L[0] = mkLayer(40, 0); spa.L[1] = mkLayer(63, 1);
  k_tf<<<Bn, 256, 0, stream>>>(featB, spa, 1, pspa);

  // ---- spectral branch ----
  k_conv0<<<Bn, 256, 0, stream>>>(X, 200, 1, 224, F(14), F(13),
                                  spe0t, F(88), F(87), F(86), featA);
  k_ms<<<Bn, 256, 0, stream>>>(featA,
      se_c1, F(100), F(93), F(92),
      se_c3, F(102), F(95), F(94),
      se_c5, F(104), F(97), F(96),
      se_pj, F(106), F(99), F(98),
      featB);
  TFP spe; spe.lnb = F(90); spe.lng = F(91); spe.pos = F(108);
  spe.L[0] = mkLayer(109, 2); spe.L[1] = mkLayer(132, 3);
  k_tf<<<Bn, 256, 0, stream>>>(featB, spe, 0, pspe);

  // ---- fused head ----
  k_head<<<Bn/32, 256, 0, stream>>>(pspa, pspe,
      fu1t, F(9), F(6), F(5),
      fu2t, F(11), F(8), F(7),
      cl1t, F(1), cl2t, F(3),
      (float*)d_out);

  (void)in_sizes; (void)n_in; (void)out_size; (void)ws_size;
}
